// VocabParallelEmbeddingWithLoRA_13030930776753
// MI455X (gfx1250) — matrix operand order repaired and checked by reading, not compiled, not run on hardware
//
#include <hip/hip_runtime.h>

// ---------------------------------------------------------------------------
// VocabParallelEmbeddingWithLoRA for MI455X (gfx1250, wave32, WMMA).
//
// out[t,d] = base_weight[base_idx[t], d] + sum_r a[t,r] * lora_b[li[t], d, r]
//
// GEMM-ified: A'[t, li[t]*16+r] = a[t,r] (zero elsewhere, K=128), and
// B'[l*16+r, d] = lora_b[l, d, r]  ->  out_delta = A'(Tx128) x B'(128x2048).
// K=128 => 4x v_wmma_f32_16x16x32_bf16 per 16x16 output tile.
// Bandwidth-bound (~70MB @ 23.3 TB/s); B' (512KB bf16) is L2-resident.
//
// Round-2 changes (scheduling only):
//  * All 4 B fragments get distinct registers -> loads issue as one batch
//    ahead of the WMMA chain (partial s_wait_loadcnt instead of 0x0 each).
//  * Base-row gathers issue before the WMMA chain to hide under matrix ops.
//  * Row/out pointers hoisted; counted column loop (static trip count 4).
// ---------------------------------------------------------------------------

typedef __attribute__((ext_vector_type(16))) __bf16 v16bf;
typedef __attribute__((ext_vector_type(8)))  __bf16 v8bf;
typedef __attribute__((ext_vector_type(8)))  float  v8f;

namespace {
constexpr int kOrgVocab = 32000;
constexpr int kExtra    = 256;
constexpr int kVocabExt = kOrgVocab + kExtra;   // 32256
constexpr int kNumLoras = 8;
constexpr int kRank     = 16;
constexpr int kD        = 2048;
constexpr int kKTot     = kNumLoras * kRank;    // 128
constexpr int kTokens   = 2 * 2048;             // B*S = 4096
constexpr int kColBlocks = 4;                   // column blocks in grid.x
constexpr int kColTilesPerBlock = (kD / 16) / kColBlocks;  // 32
constexpr int kColIters = kColTilesPerBlock / 8;           // 4 per wave
}

// float -> bf16 bits, round-to-nearest-even
__device__ __forceinline__ unsigned short f2bf(float f) {
    unsigned int u = __float_as_uint(f);
    u += 0x7FFFu + ((u >> 16) & 1u);
    return (unsigned short)(u >> 16);
}

// Transcode lora_b_stacked [8,2048,16] f32 -> Bt bf16 laid out [d][k=l*16+r],
// so each WMMA B fragment (16 contiguous K per lane) is a contiguous 32B load.
__global__ __launch_bounds__(256) void prep_bt_kernel(
        const float* __restrict__ lora_b, unsigned short* __restrict__ bt) {
    int tid = blockIdx.x * 256 + threadIdx.x;       // 2^18 total elements
    if (tid >= kNumLoras * kD * kRank) return;
    int r = tid & 15;              // rank
    int d = (tid >> 4) & (kD - 1); // column
    int l = tid >> 15;             // lora
    bt[(size_t)d * kKTot + l * kRank + r] = f2bf(lora_b[tid]);
}

__global__ __launch_bounds__(256) void lora_embed_kernel(
        const int* __restrict__ xf,            // [T] token ids
        const int* __restrict__ li,            // [T] lora indices
        const float* __restrict__ basew,       // [34048, 2048]
        const float* __restrict__ a2d,         // [8*32256, 16]
        const unsigned short* __restrict__ bt, // [2048, 128] bf16
        float* __restrict__ out) {             // [T, 2048]
    __shared__ __align__(32) unsigned short A_lds[16 * kKTot]; // 16 tok x 128 K
    __shared__ int base_row[16];

    const int tt  = blockIdx.y;     // token tile (16 tokens)
    const int tid = threadIdx.x;

    // ---- Build zero-padded bf16 A' tile in LDS --------------------------
    unsigned int* az = (unsigned int*)A_lds;      // 1024 dwords
    #pragma unroll
    for (int i = 0; i < 4; ++i) az[tid + 256 * i] = 0u;
    __syncthreads();

    {
        const int t = tid >> 4, r = tid & 15;     // (token-in-tile, rank)
        const int g = tt * 16 + t;
        const int id = xf[g];
        const int l  = li[g];
        const float av = a2d[(l * kVocabExt + id) * kRank + r];
        A_lds[t * kKTot + l * kRank + r] = f2bf(av);
        if (r == 0)
            base_row[t] = id + ((id >= kOrgVocab) ? l * kExtra : 0);
    }
    __syncthreads();

    // ---- Per-wave WMMA over column tiles --------------------------------
    const int lane = tid & 31;
    const int wave = tid >> 5;
    const int m  = lane & 15;   // A row / B col / C col within tile
    const int kh = lane >> 4;   // lane-half selector

    // A fragments: per lane 16 contiguous K at K = s*32 + kh*16 (matches B),
    // loaded as two 8-element chunks (each 16B aligned in LDS).
    union AFrag { v16bf v; v8bf h[2]; };
    AFrag afrag[4];
    #pragma unroll
    for (int s = 0; s < 4; ++s) {
        const unsigned short* base = &A_lds[m * kKTot + s * 32 + kh * 16];
        afrag[s].h[0] = *(const v8bf*)(base);
        afrag[s].h[1] = *(const v8bf*)(base + 8);
    }

    // C/D layout: VGPR v holds row M = v + 8*kh, column N = lane%16.
    // Hoist loop-invariant row pointers (base gather) and output pointers.
    const float* rowp[8];
    float*       outp[8];
    #pragma unroll
    for (int v = 0; v < 8; ++v) {
        rowp[v] = basew + (size_t)base_row[v + 8 * kh] * kD;
        outp[v] = out + (size_t)(tt * 16 + v + 8 * kh) * kD;
    }

    const int col0 = blockIdx.x * kColTilesPerBlock * 16 + wave * 16 + m;
    #pragma unroll
    for (int i = 0; i < kColIters; ++i) {
        const int col = col0 + i * 128;          // 8 waves x 16 cols apart

        // Batch-issue all 4 B fragments (distinct regs -> overlapped loads).
        const unsigned short* bcol = &bt[(size_t)col * kKTot + kh * 16];
        v16bf bf0 = *(const v16bf*)(bcol + 0 * 32);
        v16bf bf1 = *(const v16bf*)(bcol + 1 * 32);
        v16bf bf2 = *(const v16bf*)(bcol + 2 * 32);
        v16bf bf3 = *(const v16bf*)(bcol + 3 * 32);

        // Issue the 8 base-row gathers now so they hide under the WMMA chain
        // (64B coalesced per half-wave per row).
        float basev[8];
        #pragma unroll
        for (int v = 0; v < 8; ++v) basev[v] = rowp[v][col];

        v8f c = {};
        c = __builtin_amdgcn_wmma_f32_16x16x32_bf16(false, afrag[0].v, false,
                                                    bf0, (short)0, c, false, false);
        c = __builtin_amdgcn_wmma_f32_16x16x32_bf16(false, afrag[1].v, false,
                                                    bf1, (short)0, c, false, false);
        c = __builtin_amdgcn_wmma_f32_16x16x32_bf16(false, afrag[2].v, false,
                                                    bf2, (short)0, c, false, false);
        c = __builtin_amdgcn_wmma_f32_16x16x32_bf16(false, afrag[3].v, false,
                                                    bf3, (short)0, c, false, false);

        #pragma unroll
        for (int v = 0; v < 8; ++v) outp[v][col] = basev[v] + c[v];
    }
}

extern "C" void kernel_launch(void* const* d_in, const int* in_sizes, int n_in,
                              void* d_out, int out_size, void* d_ws, size_t ws_size,
                              hipStream_t stream) {
    const int*   x     = (const int*)d_in[0];     // [2,2048] int32 token ids
    const int*   lidx  = (const int*)d_in[1];     // [4096] int32
    const float* basew = (const float*)d_in[2];   // [34048,2048] f32
    const float* a2d   = (const float*)d_in[3];   // [8,32256,16] f32
    const float* lb    = (const float*)d_in[4];   // [8,2048,16] f32
    float*       out   = (float*)d_out;
    unsigned short* bt = (unsigned short*)d_ws;   // 512 KB bf16 B' table

    (void)in_sizes; (void)n_in; (void)out_size; (void)ws_size;

    // Pass 1: transcode lora_b -> bf16 B' in workspace.
    prep_bt_kernel<<<dim3((kNumLoras * kD * kRank + 255) / 256), dim3(256), 0,
                     stream>>>(lb, bt);

    // Pass 2: fused gather + WMMA LoRA GEMM + store.
    // grid: 4 column blocks x 256 token tiles; 256 threads = 8 waves.
    lora_embed_kernel<<<dim3(kColBlocks, kTokens / 16), dim3(256), 0, stream>>>(
        x, lidx, basew, a2d, bt, out);
}
